// PyTorchMoELayer_81973745811690
// MI455X (gfx1250) — matrix operand order repaired and checked by reading, not compiled, not run on hardware
//
#include <hip/hip_runtime.h>

// ---------------------------------------------------------------------------
// MoE (top-2 of 8 experts), B*S=8192 tokens, H=1024, I=2048.
// bf16 WMMA (v_wmma_f32_16x16x32_bf16), fp32 accumulate, routed compute.
// All GEMM operands pre-swizzled into WMMA fragment-native layout so every
// fragment is a contiguous 32B-per-lane load. GEMM1 uses a distance-2
// software pipeline (three register sets) pinned with sched_group_barrier;
// __launch_bounds__(256,2) sizes the VGPR budget for the true occupancy
// (2 waves/SIMD) so the pipeline holds without scratch spills.
// Workspace requirement: ~267 MiB.
// ---------------------------------------------------------------------------

typedef __attribute__((ext_vector_type(16))) __bf16 v16bf;
typedef __attribute__((ext_vector_type(8)))  float  v8f;

#define TOK   8192      // tokens (B*S)
#define HD    1024      // hidden
#define NE    8         // experts
#define ID    2048      // intermediate
#define N1    4096      // 2*ID (gate|up)
#define CAP   17408     // max padded slots: 16384 + 8*127 -> round to 128
#define CAPT  1088      // CAP/16 row-tiles

#if __has_builtin(__builtin_amdgcn_sched_group_barrier)
#define SGB(mask, size, id) __builtin_amdgcn_sched_group_barrier(mask, size, id)
#else
#define SGB(mask, size, id)
#endif
// masks: 0x008 MFMA/WMMA, 0x020 VMEM read, 0x100 DS read

// ---- fp32 -> bf16 round-to-nearest-even ----
__device__ __forceinline__ unsigned short f2bf(float f) {
  unsigned u = __builtin_bit_cast(unsigned, f);
  unsigned r = u + 0x7fffu + ((u >> 16) & 1u);
  return (unsigned short)(r >> 16);
}

// ---- WMMA 16x16x32 bf16 fragment index maps (wave32, per ISA 7.12.2) ----
__device__ __forceinline__ int a_k(int lane, int i) {
  int r = i >> 1, b = i & 1;
  return ((r >= 4) ? 16 : 0) + ((lane >> 4) << 3) + ((r & 3) << 1) + b;
}
__device__ __forceinline__ int b_k(int lane, int i) {
  int r = i >> 1, b = i & 1;
  return ((r >= 4) ? 16 : 0) + ((lane >> 4) << 3) + ((r & 3) << 1) + b;
}
__device__ __forceinline__ int afrag_pos(int m, int k) {
  int lane = (m & 15) | (((k >> 3) & 1) << 4);
  int r = ((k & 16) ? 4 : 0) + ((k & 7) >> 1);
  return lane * 16 + (r << 1) + (k & 1);
}

__device__ __forceinline__ v8f wmma_bf16(v16bf a, v16bf b, v8f c) {
  return __builtin_amdgcn_wmma_f32_16x16x32_bf16(false, a, false, b, (short)0, c,
                                                 false, false);
}

// ---------------------------------------------------------------------------
// 0) zero counters + slot arrays (pad slots read token 0 with weight 0)
__global__ void k_init(int* counts, int* slotTok, float* slotW) {
  int gid = blockIdx.x * 256 + threadIdx.x;
  if (gid < NE) counts[gid] = 0;
  if (gid < CAP) { slotTok[gid] = 0; slotW[gid] = 0.f; }
}

// 1) router: one wave per token -> top-2 experts + normalized weights
__global__ void k_router(const float* __restrict__ x, const float* __restrict__ Wr,
                         int* __restrict__ sel, float* __restrict__ wts,
                         int* __restrict__ counts) {
  int wv = threadIdx.x >> 5, lane = threadIdx.x & 31;
  int t = blockIdx.x * 8 + wv;
  float acc[NE];
#pragma unroll
  for (int e = 0; e < NE; e++) acc[e] = 0.f;
  const float* xr = x + (size_t)t * HD;
  for (int h = lane; h < HD; h += 32) {
    float xv = xr[h];
    const float* wr = Wr + h * NE;
#pragma unroll
    for (int e = 0; e < NE; e++) acc[e] += xv * wr[e];
  }
#pragma unroll
  for (int off = 16; off > 0; off >>= 1) {
#pragma unroll
    for (int e = 0; e < NE; e++) acc[e] += __shfl_xor(acc[e], off, 32);
  }
  if (lane == 0) {
    int i0 = 0; float v0 = acc[0];
#pragma unroll
    for (int e = 1; e < NE; e++) if (acc[e] > v0) { v0 = acc[e]; i0 = e; }
    int i1 = 0; float v1 = -3.4e38f;
#pragma unroll
    for (int e = 0; e < NE; e++) if (e != i0 && acc[e] > v1) { v1 = acc[e]; i1 = e; }
    float e1 = __expf(v1 - v0);           // softmax denom cancels in rw norm
    float w0 = 1.f / (1.f + e1), w1 = e1 / (1.f + e1);
    sel[2 * t] = i0; sel[2 * t + 1] = i1;
    wts[2 * t] = w0; wts[2 * t + 1] = w1;
    atomicAdd(&counts[i0], 1);
    atomicAdd(&counts[i1], 1);
  }
}

// 2) prefix over 8 experts, pad each region to 128 rows
__global__ void k_scan(const int* counts, int* padOff, int* padCnt, int* cursor) {
  if (blockIdx.x == 0 && threadIdx.x == 0) {
    int off = 0;
    for (int e = 0; e < NE; e++) {
      padOff[e] = off;
      int pc = (counts[e] + 127) & ~127;
      padCnt[e] = pc;
      cursor[e] = off;
      off += pc;
    }
  }
}

// 3) scatter tokens into expert slot lists (order-independent final result)
__global__ void k_fill(const int* sel, const float* wts, int* cursor,
                       int* slotTok, float* slotW, int* tokSlot) {
  int t = blockIdx.x * 256 + threadIdx.x;
  if (t >= TOK) return;
#pragma unroll
  for (int j = 0; j < 2; j++) {
    int e = sel[2 * t + j];
    int s = atomicAdd(&cursor[e], 1);
    slotTok[s] = t; slotW[s] = wts[2 * t + j]; tokSlot[2 * t + j] = s;
  }
}

// 4) gather x rows into Xg: [slotTile][kc(32)][512] in A-fragment order, bf16
__global__ void k_gatherx(const float* __restrict__ x, const int* __restrict__ slotTok,
                          unsigned short* __restrict__ Xg) {
  int gid = blockIdx.x * 256 + threadIdx.x;   // CAPT*32*512
  int pos = gid & 511;
  int kc  = (gid >> 9) & 31;
  int st  = gid >> 14;
  int lane = pos >> 4, i = pos & 15;
  int m = lane & 15;
  int k = a_k(lane, i);
  int tok = slotTok[st * 16 + m];
  Xg[gid] = f2bf(x[(size_t)tok * HD + kc * 32 + k]);
}

// 5) W1 fp32 (E,H,2I) -> W1s bf16 [e][ntile(256)][kc(32)][512] B-fragment order
__global__ void k_convw1(const float* __restrict__ W1, unsigned short* __restrict__ W1s) {
  int gid = blockIdx.x * 256 + threadIdx.x;   // 8*256*32*512
  int pos = gid & 511;
  int kc  = (gid >> 9) & 31;
  int nt  = (gid >> 14) & 255;
  int e   = gid >> 22;
  int lane = pos >> 4, i = pos & 15;
  int n = lane & 15;
  int k = b_k(lane, i);
  W1s[gid] = f2bf(W1[((size_t)e * HD + kc * 32 + k) * N1 + nt * 16 + n]);
}

// 6) W2 fp32 (E,I,H) -> W2s bf16 [e][ntile(64)][kc(64)][512] B-fragment order
__global__ void k_convw2(const float* __restrict__ W2, unsigned short* __restrict__ W2s) {
  int gid = blockIdx.x * 256 + threadIdx.x;   // 8*64*64*512
  int pos = gid & 511;
  int kc  = (gid >> 9) & 63;
  int nt  = (gid >> 15) & 63;
  int e   = gid >> 21;
  int lane = pos >> 4, i = pos & 15;
  int n = lane & 15;
  int k = b_k(lane, i);
  W2s[gid] = f2bf(W2[((size_t)e * ID + kc * 32 + k) * HD + nt * 16 + n]);
}

// 7) GEMM1 + SiLU*up fusion. Block = 128 rows of one expert; full X tile
//    (256KB) persisted in LDS; W1 fragments streamed from L2 with a
//    distance-2 (three register set) pipeline pinned by sched_group_barrier.
__global__ void __launch_bounds__(256, 2)
k_gemm1(const unsigned short* __restrict__ Xg,
        const unsigned short* __restrict__ W1s,
        const int* __restrict__ padOff, const int* __restrict__ padCnt,
        unsigned short* __restrict__ Hbuf) {
  int e = blockIdx.x >> 6, bt = blockIdx.x & 63;
  if (bt * 128 >= padCnt[e]) return;
  int tile0 = (padOff[e] >> 4) + bt * 8;

  extern __shared__ char smemraw[];
  unsigned short* sA = (unsigned short*)smemraw;      // 8*32*512*2B = 256KB
  {
    const uint4* src = (const uint4*)(Xg + (size_t)tile0 * 32 * 512);
    uint4* dst = (uint4*)sA;
    for (int it = threadIdx.x; it < 32768; it += 256) dst[it] = src[it];
  }
  __syncthreads();

  const v16bf* sAv = (const v16bf*)sA;
  const v16bf* Bv  = (const v16bf*)W1s;
  int wv = threadIdx.x >> 5, lane = threadIdx.x & 31;
  int hi = lane >> 4, n = lane & 15;

  for (int item = wv; item < 512; item += 8) {       // 8 M-tiles x 64 act-chunks
    int mt = item >> 6, ch = item & 63;              // chunk = 32 act cols = GEMM2 kc
    v8f ag0 = {0,0,0,0,0,0,0,0}, ag1 = {0,0,0,0,0,0,0,0};
    v8f au0 = {0,0,0,0,0,0,0,0}, au1 = {0,0,0,0,0,0,0,0};
    const v16bf* ap  = sAv + (size_t)(mt * 32) * 32 + lane;
    const v16bf* bg0 = Bv + (size_t)(e * 256 + (ch * 2    )) * 32 * 32 + lane;
    const v16bf* bg1 = Bv + (size_t)(e * 256 + (ch * 2 + 1)) * 32 * 32 + lane;
    const v16bf* bu0 = Bv + (size_t)(e * 256 + (128 + ch * 2    )) * 32 * 32 + lane;
    const v16bf* bu1 = Bv + (size_t)(e * 256 + (128 + ch * 2 + 1)) * 32 * 32 + lane;

    // distance-2 software pipeline: stage0 = kc, stage1 = kc+1; load kc+2.
    v16bf a0  = ap[0],      a1  = ap[32];
    v16bf g0a = bg0[0],     g0b = bg0[32];
    v16bf g1a = bg1[0],     g1b = bg1[32];
    v16bf u0a = bu0[0],     u0b = bu0[32];
    v16bf u1a = bu1[0],     u1b = bu1[32];
#pragma unroll 4
    for (int kc = 0; kc < 32; kc++) {
      int nk = (kc + 2 < 32) ? (kc + 2) : 31;        // clamped redundant tail load
      int no = nk * 32;
      v16bf a2  = ap[no];
      v16bf g0c = bg0[no], g1c = bg1[no], u0c = bu0[no], u1c = bu1[no];
      ag0 = wmma_bf16(a0, g0a, ag0);
      ag1 = wmma_bf16(a0, g1a, ag1);
      au0 = wmma_bf16(a0, u0a, au0);
      au1 = wmma_bf16(a0, u1a, au1);
      // pin the pipeline shape: [2 DS-read][8 VMEM-read][4 WMMA] per k-step
      SGB(0x100, 2, 0);
      SGB(0x020, 8, 0);
      SGB(0x008, 4, 0);
      a0 = a1;  a1 = a2;
      g0a = g0b; g0b = g0c;
      g1a = g1b; g1b = g1c;
      u0a = u0b; u0b = u0c;
      u1a = u1b; u1b = u1c;
    }

    unsigned short* hb = Hbuf + ((size_t)(tile0 + mt) * 64 + ch) * 512;
#pragma unroll
    for (int r = 0; r < 8; r++) {
      int m = r + (hi << 3);
      float g = ag0[r];
      float v = (g / (1.f + __expf(-g))) * au0[r];
      hb[afrag_pos(m, n)] = f2bf(v);
      g = ag1[r];
      v = (g / (1.f + __expf(-g))) * au1[r];
      hb[afrag_pos(m, 16 + n)] = f2bf(v);
    }
  }
}

// 8) GEMM2: act(128x2048) @ W2(2048x1024), scale by routing weight.
//    B double-buffered per k-step; A rolls a one-fragment prefetch across the
//    mt loop (all 8 waves share the A stream -> L0-hot). Low register use.
__global__ void __launch_bounds__(256, 2)
k_gemm2(const unsigned short* __restrict__ Hbuf,
        const unsigned short* __restrict__ W2s,
        const int* __restrict__ padOff, const int* __restrict__ padCnt,
        const float* __restrict__ slotW,
        float* __restrict__ Ybuf) {
  int e = blockIdx.x >> 6, bt = blockIdx.x & 63;
  if (bt * 128 >= padCnt[e]) return;
  int tile0 = (padOff[e] >> 4) + bt * 8;
  int wv = threadIdx.x >> 5, lane = threadIdx.x & 31;
  int hi = lane >> 4, n = lane & 15;
  const v16bf* Av = (const v16bf*)Hbuf;
  const v16bf* Bv = (const v16bf*)W2s;
  const v16bf* apBase = Av + (size_t)tile0 * 64 * 32 + lane;  // +(mt*64+kc)*32

  for (int panel = 0; panel < 8; panel++) {
    int nt = panel * 8 + wv;                          // N-tile 0..63
    v8f acc[8];
#pragma unroll
    for (int mt = 0; mt < 8; mt++) acc[mt] = (v8f){0,0,0,0,0,0,0,0};
    const v16bf* bp = Bv + (size_t)(e * 64 + nt) * 64 * 32 + lane;

    v16bf b_c = bp[0];
    v16bf a_c = apBase[0];                            // (mt=0, kc=0)
#pragma unroll 2
    for (int kc = 0; kc < 64; kc++) {
      int nk = (kc + 1 < 64) ? (kc + 1) : 63;        // clamped redundant tail load
      v16bf b_n = bp[nk * 32];
#pragma unroll
      for (int mt = 0; mt < 8; mt++) {
        int nmt = (mt + 1) & 7;
        int akc = (mt == 7) ? nk : kc;               // wrap into next k-step
        v16bf a_n = apBase[(nmt * 64 + akc) * 32];
        acc[mt] = wmma_bf16(a_c, b_c, acc[mt]);
        SGB(0x020, 2, 0);                            // [2 VMEM-read][1 WMMA]
        SGB(0x008, 1, 0);
        a_c = a_n;
      }
      b_c = b_n;
    }

#pragma unroll
    for (int mt = 0; mt < 8; mt++) {
      int st = tile0 + mt;
#pragma unroll
      for (int r = 0; r < 8; r++) {
        int row = st * 16 + r + (hi << 3);
        float wgt = slotW[row];
        Ybuf[(size_t)row * HD + nt * 16 + n] = acc[mt][r] * wgt;
      }
    }
  }
}

// 9) combine: out[t] = Y[slot0(t)] + Y[slot1(t)]  (weights already applied)
__global__ void k_combine(const float* __restrict__ Ybuf, const int* __restrict__ tokSlot,
                          float* __restrict__ out) {
  int gid = blockIdx.x * 256 + threadIdx.x;           // TOK*HD
  int t = gid >> 10, c = gid & 1023;
  int s0 = tokSlot[2 * t], s1 = tokSlot[2 * t + 1];
  out[gid] = Ybuf[(size_t)s0 * HD + c] + Ybuf[(size_t)s1 * HD + c];
}

// ---------------------------------------------------------------------------
extern "C" void kernel_launch(void* const* d_in, const int* in_sizes, int n_in,
                              void* d_out, int out_size, void* d_ws, size_t ws_size,
                              hipStream_t stream) {
  (void)in_sizes; (void)n_in; (void)out_size; (void)ws_size;
  const float* x  = (const float*)d_in[0];
  const float* Wr = (const float*)d_in[1];
  const float* W1 = (const float*)d_in[2];
  const float* W2 = (const float*)d_in[3];
  float* out = (float*)d_out;

  // workspace carve-out (512B aligned); total ~267 MiB
  char* p = (char*)d_ws;
  auto take = [&](size_t bytes) -> char* {
    char* r = p; p += (bytes + 511) & ~(size_t)511; return r;
  };
  int*   sel     = (int*)take((size_t)TOK * 2 * 4);
  float* wts     = (float*)take((size_t)TOK * 2 * 4);
  int*   counts  = (int*)take(64);
  int*   padOff  = (int*)take(64);
  int*   padCnt  = (int*)take(64);
  int*   cursor  = (int*)take(64);
  int*   slotTok = (int*)take((size_t)CAP * 4);
  float* slotW   = (float*)take((size_t)CAP * 4);
  int*   tokSlot = (int*)take((size_t)TOK * 2 * 4);
  unsigned short* Xg   = (unsigned short*)take((size_t)CAPT * 32 * 512 * 2);
  unsigned short* W1s  = (unsigned short*)take((size_t)NE * 256 * 32 * 512 * 2);
  unsigned short* W2s  = (unsigned short*)take((size_t)NE * 64 * 64 * 512 * 2);
  unsigned short* Hbuf = (unsigned short*)take((size_t)CAPT * 64 * 512 * 2);
  float*          Ybuf = (float*)take((size_t)CAP * HD * 4);

  k_init   <<<68, 256, 0, stream>>>(counts, slotTok, slotW);
  k_router <<<1024, 256, 0, stream>>>(x, Wr, sel, wts, counts);
  k_scan   <<<1, 64, 0, stream>>>(counts, padOff, padCnt, cursor);
  k_fill   <<<32, 256, 0, stream>>>(sel, wts, cursor, slotTok, slotW, tokSlot);
  k_gatherx<<<(CAPT * 32 * 512) / 256, 256, 0, stream>>>(x, slotTok, Xg);
  k_convw1 <<<(NE * 256 * 32 * 512) / 256, 256, 0, stream>>>(W1, W1s);
  k_convw2 <<<(NE * 64 * 64 * 512) / 256, 256, 0, stream>>>(W2, W2s);
  k_gemm1  <<<512, 256, 262144, stream>>>(Xg, W1s, padOff, padCnt, Hbuf);
  k_gemm2  <<<512, 256, 0, stream>>>(Hbuf, W2s, padOff, padCnt, slotW, Ybuf);
  k_combine<<<(TOK * HD) / 256, 256, 0, stream>>>(Ybuf, tokSlot, out);
}
